// GSESNN_29935922053455
// MI455X (gfx1250) — matrix operand order repaired and checked by reading, not compiled, not run on hardware
//
#include <hip/hip_runtime.h>
#include <stdint.h>

typedef __attribute__((ext_vector_type(16))) _Float16 v16h;
typedef __attribute__((ext_vector_type(8)))  float    v8f;

namespace {
constexpr int GN    = 245760;   // graph nodes  (B * N_PER)
constexpr int GE    = 2000000;  // edges
constexpr int NB    = 4096;     // batch
constexpr int NPER  = 60;
constexpr int KTOP  = 30;
constexpr int NDRUG = 1024;
constexpr int NDIS  = 2048;
}

// ======================= elementwise / graph utilities =======================

__global__ void fill_f32(float* p, float v, long n) {
  long i = (long)blockIdx.x * blockDim.x + threadIdx.x;
  if (i < n) p[i] = v;
}

__global__ void count_deg(const int* __restrict__ dst, float* __restrict__ deg, int e) {
  int i = blockIdx.x * blockDim.x + threadIdx.x;
  if (i < e) atomicAdd(&deg[dst[i]], 1.0f);
}

__global__ void rsqrt_inplace(float* p, long n) {
  long i = (long)blockIdx.x * blockDim.x + threadIdx.x;
  if (i < n) p[i] = rsqrtf(p[i]);
}

// acc[n, off+f] = h[n,f]*dinv[n]^2 + bias[f]   (self term + bias of gcn_conv)
__global__ void gcn_self_init(const float* __restrict__ h, const float* __restrict__ dinv,
                              const float* __restrict__ bias, float* __restrict__ acc,
                              int off, long n32) {
  long i = (long)blockIdx.x * blockDim.x + threadIdx.x;
  if (i >= n32) return;
  long n = i >> 5; int f = (int)(i & 31);
  float dv = dinv[n];
  acc[n * 64 + off + f] = h[i] * dv * dv + bias[f];
}

// wave-per-edge, lane = feature (32 features == wave32). Bandwidth-bound phase:
// coalesced 128B gather of h[src], f32 atomic scatter into acc[dst].
__global__ void edge_scatter(const int* __restrict__ src, const int* __restrict__ dst,
                             const float* __restrict__ h, const float* __restrict__ dinv,
                             float* __restrict__ acc, int off, int e) {
  const int lane = threadIdx.x & 31;
  const int wv   = blockIdx.x * (blockDim.x >> 5) + (threadIdx.x >> 5);
  const int nwv  = gridDim.x * (blockDim.x >> 5);
  for (int idx = wv; idx < e; idx += nwv) {
    int s = src[idx], d = dst[idx];
    if (idx + nwv < e)  // hide gather latency for the next edge this wave owns
      __builtin_prefetch(&h[(long)src[idx + nwv] * 32 + lane], 0, 3);
    float w = dinv[s] * dinv[d];
    atomicAdd(&acc[(long)d * 64 + off + lane], h[(long)s * 32 + lane] * w);
  }
}

__global__ void relu_strided(float* __restrict__ p, long rows, int ld, int off) {
  long i = (long)blockIdx.x * blockDim.x + threadIdx.x;  // rows*32 threads
  if (i >= rows * 32) return;
  long r = i >> 5; int c = (int)(i & 31);
  float* q = p + r * ld + off + c;
  *q = fmaxf(*q, 0.0f);
}

// ======================= WMMA GEMM (f16 in, f32 accumulate) ==================
// One 16 x (16*NT) output strip per wave32: the A fragment is loaded once per
// K-step and reused for NT back-to-back v_wmma issues (cuts A traffic NT x).
// Fragment layouts used here (wave32; A and B share contiguous k order):
//  A (16x32 f16): lane<16 -> K {0..15}; lane>=16 -> K {16..31}
//  B (32x16 f16): lane<16 -> K {0..15};        lane>=16 -> K {16..31}
//  C/D (16x16 f32): vgpr r -> row r + 8*(lane>=16), col = lane&15
// ACT: 0 = none, 1 = relu.  BT: B stored N x K (transposed) vs K x N.

template<int ACT, bool BT, int NT>
__global__ void __launch_bounds__(256)
gemm_wmma(const float* __restrict__ A, int lda,
          const float* __restrict__ Bm, int ldb,
          float* __restrict__ C, int ldc,
          const float* __restrict__ bias,
          int M, int Nc, int K)
{
  const int lane   = threadIdx.x & 31;
  const int half   = lane >> 4;
  const int strips = Nc / (16 * NT);
  const int tile   = blockIdx.x * (blockDim.x >> 5) + (threadIdx.x >> 5);
  if (tile >= (M >> 4) * strips) return;
  const int tm    = tile / strips;
  const int ts    = tile - tm * strips;
  const int arow  = tm * 16 + (lane & 15);
  const int bcol0 = ts * (16 * NT) + (lane & 15);

  const float* ap = A + (long)arow * lda;
  v8f acc[NT] = {};
  for (int k0 = 0; k0 < K; k0 += 32) {
    v16h af;
    const float* a0 = ap + k0 + half * 16;
#pragma unroll
    for (int j = 0; j < 8; ++j) {
      af[j]     = (_Float16)a0[j];
      af[j + 8] = (_Float16)a0[j + 8];
    }
    const int kb = k0 + half * 16;
#pragma unroll
    for (int t = 0; t < NT; ++t) {
      v16h bf;
      const int bc = bcol0 + 16 * t;
#pragma unroll
      for (int e2 = 0; e2 < 16; ++e2) {
        float v = BT ? Bm[(long)bc * ldb + kb + e2]
                     : Bm[(long)(kb + e2) * ldb + bc];
        bf[e2] = (_Float16)v;
      }
      acc[t] = __builtin_amdgcn_wmma_f32_16x16x32_f16(false, af, false, bf,
                                                      (short)0, acc[t], false, false);
    }
  }
#pragma unroll
  for (int t = 0; t < NT; ++t) {
    const int bc = bcol0 + 16 * t;
    const float bv = bias ? bias[bc] : 0.0f;
#pragma unroll
    for (int r = 0; r < 8; ++r) {
      float v = acc[t][r] + bv;
      if (ACT == 1) v = fmaxf(v, 0.0f);
      C[(long)(tm * 16 + half * 8 + r) * ldc + bc] = v;
    }
  }
}

// conv2 as WMMA GEMM with fused im2col A-load and transposed C-store:
//  row m = (b, t),  col = out-channel o2,  k = in-channel o * 5 + tap j
//  A(m,k) = pooled[b][o][t+j] at window offset o*15 + (k%5) = 3k - 2*(k mod 5)
//  -> mod-5 tracked incrementally (no integer division in the K-loop).
//  C stored as [b][o2*11 + t] (jnp flatten order); div-by-11 done incrementally.
__global__ void __launch_bounds__(256)
gemm_conv2(const float* __restrict__ pool,   // [B][128][15]
           const float* __restrict__ w,      // [256][640]  (N x K)
           const float* __restrict__ bias,   // [256]
           float* __restrict__ out)          // [B][2816]
{
  constexpr int M = NB * 11, Nc = 256, K = 640, NT = 4;
  constexpr int strips = Nc / (16 * NT);   // 4
  const int lane = threadIdx.x & 31;
  const int half = lane >> 4;
  const int tile = blockIdx.x * (blockDim.x >> 5) + (threadIdx.x >> 5);
  if (tile >= (M / 16) * strips) return;
  const int tm = tile / strips;
  const int ts = tile - tm * strips;
  const int m  = tm * 16 + (lane & 15);
  const int bb = m / 11, tt = m - bb * 11;
  const float* ab = pool + (long)bb * (128 * 15) + tt;
  const int bcol0 = ts * (16 * NT) + (lane & 15);

  int ka0 = half * 16,     m50 = ka0 % 5;        // slot 0..7  -> K = ka0 + j
  int ka1 = half * 16 + 8, m51 = ka1 % 5;        // slot 8..15 -> K = ka1 + j

  v8f acc[NT] = {};
  for (int k0 = 0; k0 < K; k0 += 32) {
    v16h af;
    {
      int k = ka0, r = m50;
#pragma unroll
      for (int j = 0; j < 8; ++j) {
        af[j] = (_Float16)ab[3 * k - 2 * r];
        ++k; if (++r == 5) r = 0;
      }
      k = ka1; r = m51;
#pragma unroll
      for (int j = 0; j < 8; ++j) {
        af[8 + j] = (_Float16)ab[3 * k - 2 * r];
        ++k; if (++r == 5) r = 0;
      }
    }
    const int kb = k0 + half * 16;
#pragma unroll
    for (int t = 0; t < NT; ++t) {
      v16h bf;
      const int bc = bcol0 + 16 * t;
#pragma unroll
      for (int e2 = 0; e2 < 16; ++e2)
        bf[e2] = (_Float16)w[(long)bc * K + kb + e2];
      acc[t] = __builtin_amdgcn_wmma_f32_16x16x32_f16(false, af, false, bf,
                                                      (short)0, acc[t], false, false);
    }
    ka0 += 32; m50 += 2; if (m50 >= 5) m50 -= 5;  // (k+32) mod 5 = (k mod 5 + 2) mod 5
    ka1 += 32; m51 += 2; if (m51 >= 5) m51 -= 5;
  }

  float bv[NT];
#pragma unroll
  for (int t = 0; t < NT; ++t) bv[t] = bias[bcol0 + 16 * t];

  int m0 = tm * 16 + half * 8;
  int b2 = m0 / 11, t2 = m0 - b2 * 11;            // one division, then incremental
#pragma unroll
  for (int r = 0; r < 8; ++r) {
    float* orow = out + (long)b2 * 2816 + t2;
#pragma unroll
    for (int t = 0; t < NT; ++t)
      orow[(bcol0 + 16 * t) * 11] = fmaxf(acc[t][r] + bv[t], 0.0f);
    if (++t2 == 11) { t2 = 0; ++b2; }
  }
}

// ======================= SortPool / maxpool ==================================

// Stable descending rank on the last feature; copy top-30 rows coalesced.
__global__ void sortpool(const float* __restrict__ cs, float* __restrict__ xs) {
  __shared__ float key[NPER];
  __shared__ int   sel[KTOP];
  const int b = blockIdx.x, tid = threadIdx.x;   // blockDim = 64
  if (tid < NPER) key[tid] = cs[((long)b * NPER + tid) * 64 + 63];
  __syncthreads();
  if (tid < NPER) {
    float kv = key[tid];
    int rank = 0;
    for (int j = 0; j < NPER; ++j) {
      float kj = key[j];
      rank += (kj > kv) || (kj == kv && j < tid);   // argsort(-key), stable
    }
    if (rank < KTOP) sel[rank] = tid;
  }
  __syncthreads();
  for (int i = tid; i < KTOP * 64; i += 64) {
    int kk = i >> 6, f = i & 63;
    xs[(long)b * (KTOP * 64) + i] = cs[((long)b * NPER + sel[kk]) * 64 + f];
  }
}

// conv1 output is [(b,k), o]; pool pairs of k and transpose to [b][o][t]
__global__ void maxpool2(const float* __restrict__ c1o, float* __restrict__ pool, long n) {
  long i = (long)blockIdx.x * blockDim.x + threadIdx.x;   // NB*128*15
  if (i >= n) return;
  int t = (int)(i % 15);
  int o = (int)((i / 15) % 128);
  long b = i / (15 * 128);
  float a0 = c1o[(b * 30 + 2 * t) * 128 + o];
  float a1 = c1o[(b * 30 + 2 * t + 1) * 128 + o];
  pool[i] = fmaxf(a0, a1);
}

// ======================= dense-GCN adjacency prep ============================

// deg[i] = rowsum(adj) - diag + 1 ;  dinv[i] = deg>0 ? deg^-1/2 : 0
__global__ void adj_deg(const float* __restrict__ adj, float* __restrict__ dinv, int n) {
  const int lane = threadIdx.x & 31;
  const int row  = blockIdx.x * (blockDim.x >> 5) + (threadIdx.x >> 5);
  if (row >= n) return;
  float s = 0.f;
  for (int j = lane; j < n; j += 32) s += adj[(long)row * n + j];
#pragma unroll
  for (int off = 16; off; off >>= 1) s += __shfl_xor(s, off, 32);
  if (lane == 0) {
    s = s - adj[(long)row * n + row] + 1.0f;
    dinv[row] = (s > 0.f) ? rsqrtf(s) : 0.f;
  }
}

__global__ void adj_norm(const float* __restrict__ adj, const float* __restrict__ dinv,
                         float* __restrict__ anorm, int n) {
  long i = (long)blockIdx.x * blockDim.x + threadIdx.x;
  long nn = (long)n * n;
  if (i >= nn) return;
  int r = (int)(i / n), c = (int)(i % n);
  float a = (r == c) ? 1.0f : adj[i];
  anorm[i] = dinv[r] * dinv[c] * a;
}

// ======================= fused head ==========================================
// per-wave batch element: BN + relu, p mixing, node gathers, fcs (64->32) via
// LDS broadcast, relu, fcs2 (32->1) via shuffle reduction.
__global__ void __launch_bounds__(256)
final_head(const float* __restrict__ hfin,  const float* __restrict__ bng,
           const float* __restrict__ bnb,   const float* __restrict__ dio,
           const float* __restrict__ dro,   const int* __restrict__ node,
           const float* __restrict__ pscal, const float* __restrict__ fw,
           const float* __restrict__ fb,    const float* __restrict__ f2w,
           const float* __restrict__ f2b,   float* __restrict__ out)
{
  __shared__ float zsh[8][64];
  const int lane = threadIdx.x & 31;
  const int wv   = threadIdx.x >> 5;
  const int b    = blockIdx.x * 8 + wv;          // grid = NB/8, exact
  const float p  = *pscal;
  float hb = hfin[(long)b * 32 + lane];
  hb = fmaxf(hb * bng[lane] + bnb[lane], 0.0f);
  zsh[wv][lane] = hb * p;
  int n0 = node[b * 2 + 0], n1 = node[b * 2 + 1];
  float fv = (lane < 16) ? dio[(long)n0 * 16 + lane] : dro[(long)n1 * 16 + (lane - 16)];
  zsh[wv][32 + lane] = fv * (1.0f - p);
  __syncthreads();
  float s = fb[lane];
#pragma unroll 8
  for (int i = 0; i < 64; ++i) s += zsh[wv][i] * fw[i * 32 + lane];
  s = fmaxf(s, 0.0f);
  float t = s * f2w[lane];
#pragma unroll
  for (int off = 16; off; off >>= 1) t += __shfl_xor(t, off, 32);
  if (lane == 0) out[b] = t + f2b[0];
}

// ======================= host launcher =======================================

template<int ACT, bool BT, int NT>
static void launch_gemm(const float* A, int lda, const float* Bm, int ldb,
                        float* C, int ldc, const float* bias,
                        int M, int Nc, int K, hipStream_t s) {
  int tiles  = (M / 16) * (Nc / (16 * NT));
  int blocks = (tiles + 7) / 8;
  gemm_wmma<ACT, BT, NT><<<blocks, 256, 0, s>>>(A, lda, Bm, ldb, C, ldc, bias, M, Nc, K);
}

extern "C" void kernel_launch(void* const* d_in, const int* in_sizes, int n_in,
                              void* d_out, int out_size, void* d_ws, size_t ws_size,
                              hipStream_t stream) {
  (void)in_sizes; (void)n_in; (void)out_size; (void)ws_size;
  const float* x        = (const float*)d_in[0];
  const int*   eidx     = (const int*)  d_in[1];
  const int*   node     = (const int*)  d_in[3];
  const float* di_sim   = (const float*)d_in[4];
  const float* dr_sim   = (const float*)d_in[5];
  const float* drug_adj = (const float*)d_in[6];
  const float* dis_adj  = (const float*)d_in[7];
  const float* p        = (const float*)d_in[8];
  const float* W1   = (const float*)d_in[9];  const float* b1   = (const float*)d_in[10];
  const float* W2   = (const float*)d_in[11]; const float* b2   = (const float*)d_in[12];
  const float* W3   = (const float*)d_in[13]; const float* b3   = (const float*)d_in[14];
  const float* W4   = (const float*)d_in[15]; const float* b4   = (const float*)d_in[16];
  const float* ldiw = (const float*)d_in[17]; const float* ldib = (const float*)d_in[18];
  const float* ldrw = (const float*)d_in[19]; const float* ldrb = (const float*)d_in[20];
  const float* c1w  = (const float*)d_in[21]; const float* c1b  = (const float*)d_in[22];
  const float* c2w  = (const float*)d_in[23]; const float* c2b  = (const float*)d_in[24];
  const float* l1w  = (const float*)d_in[25]; const float* l1b  = (const float*)d_in[26];
  const float* l2w  = (const float*)d_in[27]; const float* l2b  = (const float*)d_in[28];
  const float* bng  = (const float*)d_in[29]; const float* bnb  = (const float*)d_in[30];
  const float* fw   = (const float*)d_in[31]; const float* fb   = (const float*)d_in[32];
  const float* f2w  = (const float*)d_in[33]; const float* f2b  = (const float*)d_in[34];
  float* out = (float*)d_out;

  const int* src = eidx;
  const int* dst = eidx + GE;

  // ---- workspace bump allocation (floats) ----
  float* w = (float*)d_ws;
  auto alloc = [&](size_t n) { float* r = w; w += n; return r; };
  float* dinv = alloc(GN);                  // degree -> dinv (in place)
  float* hbuf = alloc((size_t)GN * 32);     // x@W (layer 1, then layer 2)
  float* cs   = alloc((size_t)GN * 64);     // [x1 | x2] accumulators
  float* xs   = alloc((size_t)NB * KTOP * 64);
  float* c1o  = cs;                         // alias: cs dead after sortpool (15.7M each)
  float* pool = xs;                         // alias: xs dead after conv1 (7.9M each)
  float* c2o  = alloc((size_t)NB * 2816);
  float* l1o  = alloc((size_t)NB * 128);
  float* hfin = alloc((size_t)NB * 32);
  float* die  = alloc((size_t)NDRUG * 256);
  float* tdi  = alloc((size_t)NDRUG * 64);
  float* dif  = alloc((size_t)NDRUG * 64);
  float* t2di = alloc((size_t)NDRUG * 16);
  float* dio  = alloc((size_t)NDRUG * 16);
  float* dre  = alloc((size_t)NDIS * 256);
  float* tdr  = alloc((size_t)NDIS * 64);
  float* drf  = alloc((size_t)NDIS * 64);
  float* t2dr = alloc((size_t)NDIS * 16);
  float* dro  = alloc((size_t)NDIS * 16);
  float* ddi  = alloc(NDRUG);
  float* ddr  = alloc(NDIS);
  float* andi = alloc((size_t)NDRUG * NDRUG);
  float* andr = alloc((size_t)NDIS * NDIS);

  // ---- big-graph normalization: deg = 1 + indeg; dinv = deg^-1/2 ----
  fill_f32<<<(GN + 255) / 256, 256, 0, stream>>>(dinv, 1.0f, GN);
  count_deg<<<(GE + 255) / 256, 256, 0, stream>>>(dst, dinv, GE);
  rsqrt_inplace<<<(GN + 255) / 256, 256, 0, stream>>>(dinv, GN);

  // ---- GCN layer 1: h1 = x@W1 ; cs[:,0:32] = scatter + self + b1 ; relu ----
  launch_gemm<0, false, 2>(x, 64, W1, 32, hbuf, 32, nullptr, GN, 32, 64, stream);
  gcn_self_init<<<(GN * 32 + 255) / 256, 256, 0, stream>>>(hbuf, dinv, b1, cs, 0, (long)GN * 32);
  edge_scatter<<<4096, 256, 0, stream>>>(src, dst, hbuf, dinv, cs, 0, GE);
  relu_strided<<<(GN * 32 + 255) / 256, 256, 0, stream>>>(cs, GN, 64, 0);

  // ---- GCN layer 2: h2 = x1@W2 ; cs[:,32:64] ; relu ----
  launch_gemm<0, false, 2>(cs, 64, W2, 32, hbuf, 32, nullptr, GN, 32, 32, stream);
  gcn_self_init<<<(GN * 32 + 255) / 256, 256, 0, stream>>>(hbuf, dinv, b2, cs, 32, (long)GN * 32);
  edge_scatter<<<4096, 256, 0, stream>>>(src, dst, hbuf, dinv, cs, 32, GE);
  relu_strided<<<(GN * 32 + 255) / 256, 256, 0, stream>>>(cs, GN, 64, 32);

  // ---- SortPool top-30 per batch -> xs (B, 30*64) ----
  sortpool<<<NB, 64, 0, stream>>>(cs, xs);

  // ---- conv1 (kernel=stride=64) == GEMM (B*30, 64) @ c1_w^T, relu ----
  launch_gemm<1, true, 4>(xs, 64, c1w, 64, c1o, 128, c1b, NB * 30, 128, 64, stream);

  // ---- maxpool k-pairs + transpose -> pooled [B][128][15] ----
  {
    long n = (long)NB * 128 * 15;
    maxpool2<<<(int)((n + 255) / 256), 256, 0, stream>>>(c1o, pool, n);
  }

  // ---- conv2 as fused-im2col WMMA GEMM -> c2o [B][2816] (channel-major) ----
  {
    int tiles = (NB * 11 / 16) * (256 / 64);
    gemm_conv2<<<(tiles + 7) / 8, 256, 0, stream>>>(pool, c2w, c2b, c2o);
  }

  // ---- lin1 (relu) and lin2 ----
  launch_gemm<1, false, 4>(c2o, 2816, l1w, 128, l1o, 128, l1b, NB, 128, 2816, stream);
  launch_gemm<0, false, 2>(l1o, 128, l2w, 32, hfin, 32, l2b, NB, 32, 128, stream);

  // ---- drug branch (n = 1024) ----
  adj_deg<<<(NDRUG + 7) / 8, 256, 0, stream>>>(drug_adj, ddi, NDRUG);
  adj_norm<<<(NDRUG * NDRUG + 255) / 256, 256, 0, stream>>>(drug_adj, ddi, andi, NDRUG);
  launch_gemm<0, false, 4>(di_sim, NDRUG, ldiw, 256, die, 256, ldib, NDRUG, 256, NDRUG, stream);
  launch_gemm<0, false, 4>(die, 256, W3, 64, tdi, 64, nullptr, NDRUG, 64, 256, stream);
  launch_gemm<1, false, 4>(andi, NDRUG, tdi, 64, dif, 64, b3, NDRUG, 64, NDRUG, stream);
  launch_gemm<0, false, 1>(dif, 64, W4, 16, t2di, 16, nullptr, NDRUG, 16, 64, stream);
  launch_gemm<1, false, 1>(andi, NDRUG, t2di, 16, dio, 16, b4, NDRUG, 16, NDRUG, stream);

  // ---- disease branch (n = 2048) ----
  adj_deg<<<(NDIS + 7) / 8, 256, 0, stream>>>(dis_adj, ddr, NDIS);
  {
    long n = (long)NDIS * NDIS;
    adj_norm<<<(int)((n + 255) / 256), 256, 0, stream>>>(dis_adj, ddr, andr, NDIS);
  }
  launch_gemm<0, false, 4>(dr_sim, NDIS, ldrw, 256, dre, 256, ldrb, NDIS, 256, NDIS, stream);
  launch_gemm<0, false, 4>(dre, 256, W3, 64, tdr, 64, nullptr, NDIS, 64, 256, stream);
  launch_gemm<1, false, 4>(andr, NDIS, tdr, 64, drf, 64, b3, NDIS, 64, NDIS, stream);
  launch_gemm<0, false, 1>(drf, 64, W4, 16, t2dr, 16, nullptr, NDIS, 16, 64, stream);
  launch_gemm<1, false, 1>(andr, NDIS, t2dr, 16, dro, 16, b4, NDIS, 16, NDIS, stream);

  // ---- fused BN + mix + gather + fcs + fcs2 head ----
  final_head<<<NB / 8, 256, 0, stream>>>(hfin, bng, bnb, dio, dro, node, p,
                                         fw, fb, f2w, f2b, out);
}
